// ResGCN_59957743452557
// MI455X (gfx1250) — matrix operand order repaired and checked by reading, not compiled, not run on hardware
//
#include <hip/hip_runtime.h>
#include <hip/hip_bf16.h>

typedef __attribute__((ext_vector_type(16))) _Float16 v16h;
typedef __attribute__((ext_vector_type(8)))  _Float16 h8;
typedef __attribute__((ext_vector_type(4)))  _Float16 h4;
typedef __attribute__((ext_vector_type(8)))  float    v8f;

#define D 128

// ---------------------------------------------------------------------------
// Convert a [128x128] fp32 weight matrix into TRANSPOSED f16 hi/lo planes:
//   hi[n*128+k] = f16(W[k][n]);  lo[n*128+k] = f16(W[k][n] - f32(hi))
// Transposed so B-fragment K-runs are contiguous 128-bit loads.
// ---------------------------------------------------------------------------
__global__ void convert_w_kernel(const float* __restrict__ W,
                                 _Float16* __restrict__ hi,
                                 _Float16* __restrict__ lo) {
    int idx = blockIdx.x * blockDim.x + threadIdx.x;      // 0 .. 16383
    if (idx >= D * D) return;
    int k = idx >> 7;
    int n = idx & (D - 1);
    float w = W[idx];                                     // W[k][n]
    _Float16 h = (_Float16)w;
    hi[n * D + k] = h;
    lo[n * D + k] = (_Float16)(w - (float)h);
}

// ---------------------------------------------------------------------------
// C[M x 128] = H[M x 128] @ W[128 x 128] (+ bias), fp32 in/out.
// fp32 emulated via f16 hi/lo split: hi*hi + lo*hi + hi*lo (3 WMMAs / chunk).
// Block = 256 threads = 8 waves; block tile = 16 rows x 128 cols; each wave
// owns one 16x16 output tile. The 16x128 A tile is converted to f16 hi/lo
// ONCE per block through LDS (8 elem/thread), then each lane reads its WMMA
// fragment slices as 16B-aligned ds_load_b128's. No divergence before WMMA,
// so EXEC is all ones as the ISA requires.
// ---------------------------------------------------------------------------
__global__ void __launch_bounds__(256)
gemm_wmma_kernel(const float*    __restrict__ H,
                 const _Float16* __restrict__ Whi,   // [n][k] transposed
                 const _Float16* __restrict__ Wlo,   // [n][k] transposed
                 const float*    __restrict__ bias,  // may be nullptr
                 float*          __restrict__ Out,
                 int nrows) {
    __shared__ _Float16 lds_hi[16 * D];   // 4 KB
    __shared__ _Float16 lds_lo[16 * D];   // 4 KB

    const int t    = threadIdx.x;
    const int lane = t & 31;
    const int wave = t >> 5;              // 0..7 -> column tile
    const int lr   = lane & 15;
    const int hh   = lane >> 4;           // half-wave select (0/1)
    const int m0   = blockIdx.x * 16;
    const int n0   = wave * 16;
    const int n    = n0 + lr;

    // ---- cooperative A-tile stage: 16x128 fp32 -> f16 hi/lo planes in LDS
#pragma unroll
    for (int r = 0; r < 2; ++r) {
        int idx = r * 256 + t;            // 0..511 float4 slots
        int m   = idx >> 5;               // 0..15
        int ko  = (idx & 31) << 2;        // 0,4,...,124
        int mrow = m0 + m;
        if (mrow > nrows - 1) mrow = nrows - 1;
        float4 xv = *(const float4*)(H + (size_t)mrow * D + ko);
        h4 hv, lv;
        hv[0] = (_Float16)xv.x;  lv[0] = (_Float16)(xv.x - (float)hv[0]);
        hv[1] = (_Float16)xv.y;  lv[1] = (_Float16)(xv.y - (float)hv[1]);
        hv[2] = (_Float16)xv.z;  lv[2] = (_Float16)(xv.z - (float)hv[2]);
        hv[3] = (_Float16)xv.w;  lv[3] = (_Float16)(xv.w - (float)hv[3]);
        *(h4*)(&lds_hi[m * D + ko]) = hv;
        *(h4*)(&lds_lo[m * D + ko]) = lv;
    }
    __syncthreads();

    const _Float16* bh = Whi + (size_t)n * D;
    const _Float16* bl = Wlo + (size_t)n * D;

    v8f acc = {};
#pragma unroll
    for (int kc = 0; kc < 4; ++kc) {
        const int k0 = kc * 32;
        v16h a_hi, a_lo, b_hi, b_lo;

        // ---- A fragment (16x32 f16), ISA layout:
        // lanes 0-15: K = {0..7, 16..23}; lanes 16-31: K = {8..15, 24..31}
        // -> two contiguous 8-half (16B) runs per plane.
        {
            const int base = lr * D + k0 + (hh << 3);
            h8 p0 = *(const h8*)(&lds_hi[base]);
            h8 p1 = *(const h8*)(&lds_hi[base + 16]);
            h8 q0 = *(const h8*)(&lds_lo[base]);
            h8 q1 = *(const h8*)(&lds_lo[base + 16]);
#pragma unroll
            for (int i = 0; i < 8; ++i) {
                a_hi[i]     = p0[i];
                a_hi[8 + i] = p1[i];
                a_lo[i]     = q0[i];
                a_lo[8 + i] = q1[i];
            }
        }

        // ---- B fragment (32x16 f16): lane lr = col N; lanes 0-15 hold
        // K = {0..7, 16..23}; lanes 16-31 hold K = {8..15, 24..31}.
        {
            const int base = k0 + (hh << 3);
            h8 p0 = *(const h8*)(&bh[base]);
            h8 p1 = *(const h8*)(&bh[base + 16]);
            h8 q0 = *(const h8*)(&bl[base]);
            h8 q1 = *(const h8*)(&bl[base + 16]);
#pragma unroll
            for (int i = 0; i < 8; ++i) {
                b_hi[i]     = p0[i];
                b_hi[8 + i] = p1[i];
                b_lo[i]     = q0[i];
                b_lo[8 + i] = q1[i];
            }
        }

        acc = __builtin_amdgcn_wmma_f32_16x16x32_f16(false, a_hi, false, b_hi,
                                                     (short)0, acc, false, false);
        acc = __builtin_amdgcn_wmma_f32_16x16x32_f16(false, a_lo, false, b_hi,
                                                     (short)0, acc, false, false);
        acc = __builtin_amdgcn_wmma_f32_16x16x32_f16(false, a_hi, false, b_lo,
                                                     (short)0, acc, false, false);
    }

    const float bv = bias ? bias[n] : 0.0f;
    // C/D layout: VGPR r -> row m0 + hh*8 + r, col n
#pragma unroll
    for (int r = 0; r < 8; ++r) {
        int m = m0 + (hh << 3) + r;
        if (m < nrows) Out[(size_t)m * D + n] = acc[r] + bv;
    }
}

// ---------------------------------------------------------------------------
// Zero-fill (float4 granularity).
// ---------------------------------------------------------------------------
__global__ void zero_kernel(float4* __restrict__ p, int total4) {
    int i = blockIdx.x * blockDim.x + threadIdx.x;
    if (i < total4) p[i] = make_float4(0.f, 0.f, 0.f, 0.f);
}

// ---------------------------------------------------------------------------
// Edge scatter-add: one wave per edge; lane d handles dims 4d..4d+3.
// agg[row[e]][:] += z[col[e]][:]   via global_atomic_add_f32 (hits in L2).
// ---------------------------------------------------------------------------
__global__ void scatter_add_kernel(const float* __restrict__ z,
                                   const int*   __restrict__ row,
                                   const int*   __restrict__ col,
                                   float*       __restrict__ agg,
                                   int E) {
    int wid  = (int)((blockIdx.x * blockDim.x + threadIdx.x) >> 5);
    int lane = threadIdx.x & 31;
    if (wid >= E) return;
    int r = row[wid];
    int c = col[wid];
    float4 v = ((const float4*)(z + (size_t)c * D))[lane];
    float* dst = agg + (size_t)r * D + lane * 4;
    atomicAdd(dst + 0, v.x);
    atomicAdd(dst + 1, v.y);
    atomicAdd(dst + 2, v.z);
    atomicAdd(dst + 3, v.w);
}

// ---------------------------------------------------------------------------
// h = h + relu(agg + b)  (float4 granularity; b is [128] -> 32 float4s)
// ---------------------------------------------------------------------------
__global__ void residual_relu_kernel(float4*       __restrict__ h,
                                     const float4* __restrict__ agg,
                                     const float*  __restrict__ bias,
                                     int total4) {
    int i = blockIdx.x * blockDim.x + threadIdx.x;
    if (i >= total4) return;
    float4 bb = ((const float4*)bias)[i & 31];
    float4 a  = agg[i];
    float4 hv = h[i];
    hv.x += fmaxf(a.x + bb.x, 0.f);
    hv.y += fmaxf(a.y + bb.y, 0.f);
    hv.z += fmaxf(a.z + bb.z, 0.f);
    hv.w += fmaxf(a.w + bb.w, 0.f);
    h[i] = hv;
}

// ---------------------------------------------------------------------------
extern "C" void kernel_launch(void* const* d_in, const int* in_sizes, int n_in,
                              void* d_out, int out_size, void* d_ws, size_t ws_size,
                              hipStream_t stream) {
    const float* x    = (const float*)d_in[0];
    const int*   edge = (const int*)  d_in[1];   // [2,E]: row then col
    const float* W_t  = (const float*)d_in[2];
    const float* b_t  = (const float*)d_in[3];
    const float* W0   = (const float*)d_in[4];
    const float* b0   = (const float*)d_in[5];
    const float* W1   = (const float*)d_in[6];
    const float* b1   = (const float*)d_in[7];

    const int Nn = in_sizes[0] / D;     // 40000 nodes
    const int E  = in_sizes[1] / 2;     // 640000 edges
    const int* row = edge;
    const int* col = edge + E;

    float* h = (float*)d_out;           // h lives in d_out throughout

    // workspace layout
    char* ws = (char*)d_ws;
    const size_t WSZ = (size_t)D * D * sizeof(_Float16);   // 32 KB per plane
    _Float16* Whi_t = (_Float16*)(ws + 0 * WSZ);
    _Float16* Wlo_t = (_Float16*)(ws + 1 * WSZ);
    _Float16* Whi0  = (_Float16*)(ws + 2 * WSZ);
    _Float16* Wlo0  = (_Float16*)(ws + 3 * WSZ);
    _Float16* Whi1  = (_Float16*)(ws + 4 * WSZ);
    _Float16* Wlo1  = (_Float16*)(ws + 5 * WSZ);
    float* z   = (float*)(ws + 6 * WSZ);
    float* agg = z + (size_t)Nn * D;

    const int total4    = Nn * (D / 4);
    const int ew_blocks = (E * 32 + 255) / 256;            // one wave per edge
    const int mtiles    = (Nn + 15) / 16;

    // 1) convert weights to transposed f16 hi/lo planes
    convert_w_kernel<<<64, 256, 0, stream>>>(W_t, Whi_t, Wlo_t);
    convert_w_kernel<<<64, 256, 0, stream>>>(W0,  Whi0,  Wlo0);
    convert_w_kernel<<<64, 256, 0, stream>>>(W1,  Whi1,  Wlo1);

    // 2) h = x @ W_t + b_t
    gemm_wmma_kernel<<<mtiles, 256, 0, stream>>>(x, Whi_t, Wlo_t, b_t, h, Nn);

    // 3) layer 0
    gemm_wmma_kernel<<<mtiles, 256, 0, stream>>>(h, Whi0, Wlo0, nullptr, z, Nn);
    zero_kernel<<<(total4 + 255) / 256, 256, 0, stream>>>((float4*)agg, total4);
    scatter_add_kernel<<<ew_blocks, 256, 0, stream>>>(z, row, col, agg, E);
    residual_relu_kernel<<<(total4 + 255) / 256, 256, 0, stream>>>(
        (float4*)h, (const float4*)agg, b0, total4);

    // 4) layer 1
    gemm_wmma_kernel<<<mtiles, 256, 0, stream>>>(h, Whi1, Wlo1, nullptr, z, Nn);
    zero_kernel<<<(total4 + 255) / 256, 256, 0, stream>>>((float4*)agg, total4);
    scatter_add_kernel<<<ew_blocks, 256, 0, stream>>>(z, row, col, agg, E);
    residual_relu_kernel<<<(total4 + 255) / 256, 256, 0, stream>>>(
        (float4*)h, (const float4*)agg, b1, total4);
}
